// RNN_47004122087506
// MI455X (gfx1250) — compile-verified, matrix operand order checked by reading, not run on hardware
//
#include <hip/hip_runtime.h>

#define B_      4
#define D_      512
#define TWOD    1024
#define L_      1024
#define NLAYERS 4
#define VOCAB   32000

#define RNN_NWG     16
#define RNN_THREADS 128   // 4 waves/WG, 64 waves total == TWOD/16 tiles

#define PROJ_THREADS 256
#define NT_PER_WAVE  4
#define MTILES   (B_ * L_ / 16)               // 256
#define NGROUPS  (VOCAB / (16 * NT_PER_WAVE)) // 500

typedef __attribute__((ext_vector_type(16))) __bf16    v16bf;
typedef __attribute__((ext_vector_type(8)))  float     v8f;
typedef __attribute__((ext_vector_type(4)))  unsigned  v4u;

union FragAB { v16bf v; v4u q[2]; };
union FragC  { v8f   v; float f[8]; };

__device__ inline unsigned short f32_to_bf16_rne(float x) {
    unsigned u = __builtin_bit_cast(unsigned, x);
    unsigned r = 0x7FFFu + ((u >> 16) & 1u);
    return (unsigned short)((u + r) >> 16);
}

__global__ void cvt_f32_to_bf16(const float* __restrict__ src,
                                unsigned short* __restrict__ dst, int n) {
    int i = blockIdx.x * blockDim.x + threadIdx.x;
    if (i < n) dst[i] = f32_to_bf16_rne(src[i]);
}

// Sbuf parity-0 slot <- starter broadcast over batch: S[0][l][b][d] = starter[l][d]
__global__ void init_state_kernel(const float* __restrict__ starter,
                                  float* __restrict__ Sbuf) {
    int i = blockIdx.x * blockDim.x + threadIdx.x;
    if (i < NLAYERS * B_ * D_) {
        int l = i / (B_ * D_);
        int d = i % D_;
        Sbuf[i] = starter[l * D_ + d];
    }
}

// Persistent cooperative RNN: 4096 sequential layer-steps with grid barriers.
__launch_bounds__(RNN_THREADS)
__global__ void rnn_kernel(const int*   __restrict__ ids,     // [B][L]
                           const float* __restrict__ embed,   // [VOCAB][D]
                           const float* __restrict__ bias,    // [NLAYERS][2D]
                           const unsigned short* __restrict__ Wb, // [NLAYERS][2D][2D] bf16
                           float* __restrict__ Xbuf,          // [2][B][D]
                           float* __restrict__ Sbuf,          // [2][NLAYERS][B][D]
                           unsigned short* __restrict__ ys,   // [B*L][D] bf16
                           unsigned* __restrict__ barrier_cnt) {
    // Full 16-row A tile, rows B_..15 permanently zero -> unconditional ds loads.
    __shared__ __align__(16) unsigned short catA[16][TWOD];

    const int tid   = threadIdx.x;
    const int lane  = tid & 31;
    const int gwave = blockIdx.x * (RNN_THREADS >> 5) + (tid >> 5);
    const int n0    = gwave * 16;          // this wave's 16 output columns

    const int arow = lane & 15;            // A-matrix row (batch index when < B_)
    const int koff = (lane >> 4) * 8;      // upper-half lanes hold K+8..K+15

    // Zero the pad rows once (rows 0..3 are restaged every layer-step).
    for (int idx = tid; idx < 16 * TWOD; idx += RNN_THREADS)
        (&catA[0][0])[idx] = 0;
    __syncthreads();

    const unsigned short* catRow = &catA[arow][0];

    for (int lc = 0; lc < L_ * NLAYERS; ++lc) {
        const int t = lc / NLAYERS;
        const int l = lc % NLAYERS;

        // ---- stage cat = [x | s] (rows 0..3 only) into LDS as bf16 ----
        const float* Sread = Sbuf + ((size_t)(t & 1) * NLAYERS + l) * (B_ * D_);
        const float* Xread = Xbuf + (size_t)((lc + 1) & 1) * (B_ * D_);
        for (int idx = tid; idx < B_ * TWOD; idx += RNN_THREADS) {
            int b = idx >> 10;             // / TWOD
            int k = idx & (TWOD - 1);
            float v;
            if (k < D_) {
                if (l == 0) {
                    int id = ids[b * L_ + t];
                    v = embed[(size_t)id * D_ + k];
                } else {
                    v = Xread[b * D_ + k];
                }
            } else {
                v = Sread[b * D_ + (k - D_)];
            }
            catA[b][k] = f32_to_bf16_rne(v);
        }
        __syncthreads();

        // ---- out[b, n0..n0+15] = cat . W[n,:]^T via bf16 WMMA, K=1024 ----
        FragC acc;
        #pragma unroll
        for (int i = 0; i < 8; ++i) acc.f[i] = 0.0f;

        const unsigned short* Wrow =
            Wb + ((size_t)l * TWOD + (n0 + (lane & 15))) * TWOD;

        // 1-deep software pipeline: prefetch k+32 fragments before k's WMMA.
        FragAB fa, fb;
        fa.q[0] = *(const v4u*)(catRow + koff);
        fa.q[1] = *(const v4u*)(catRow + 16 + koff);
        fb.q[0] = *(const v4u*)(Wrow + koff);
        fb.q[1] = *(const v4u*)(Wrow + 16 + koff);

        #pragma unroll
        for (int k0 = 0; k0 < TWOD; k0 += 32) {
            const int kn = k0 + 32;
            FragAB nfa, nfb;
            if (kn < TWOD) {
                nfa.q[0] = *(const v4u*)(catRow + kn + koff);
                nfa.q[1] = *(const v4u*)(catRow + kn + 16 + koff);
                nfb.q[0] = *(const v4u*)(Wrow + kn + koff);
                nfb.q[1] = *(const v4u*)(Wrow + kn + 16 + koff);
            }
            acc.v = __builtin_amdgcn_wmma_f32_16x16x32_bf16(
                false, fa.v, false, fb.v, (short)0, acc.v, false, false);
            if (kn < TWOD) { fa = nfa; fb = nfb; }
        }

        // ---- writeback: rows 0..3 live in elems 0..3 of lanes 0..15 ----
        if (lane < 16) {
            const int n = n0 + lane;
            const float bn = bias[l * TWOD + n];
            float* Xw = Xbuf + (size_t)(lc & 1) * (B_ * D_);
            float* Sw = Sbuf + ((size_t)((t + 1) & 1) * NLAYERS + l) * (B_ * D_);
            #pragma unroll
            for (int r = 0; r < B_; ++r) {
                float v = acc.f[r] + bn;
                if (n < D_) {
                    Xw[r * D_ + n] = v;                        // x_new -> next layer
                    if (l == NLAYERS - 1)
                        ys[((size_t)r * L_ + t) * D_ + n] = f32_to_bf16_rne(v);
                } else {
                    Sw[r * D_ + (n - D_)] = v;                 // s_new -> next step
                }
            }
        }

        // ---- grid-wide barrier (monotonic counter) ----
        __syncthreads();
        if (tid == 0) {
            __threadfence();
            atomicAdd(barrier_cnt, 1u);
            const unsigned target = (unsigned)(lc + 1) * RNN_NWG;
            while (__hip_atomic_load(barrier_cnt, __ATOMIC_RELAXED,
                                     __HIP_MEMORY_SCOPE_AGENT) < target) {
                __builtin_amdgcn_s_sleep(1);
            }
            __threadfence();
        }
        __syncthreads();
    }
}

// logits[b*L+t][v] = ys[b*L+t][:] . Wout[v][:] + bout[v]
__launch_bounds__(PROJ_THREADS)
__global__ void proj_kernel(const unsigned short* __restrict__ ys,    // [B*L][D] bf16
                            const unsigned short* __restrict__ Woutb, // [VOCAB][D] bf16
                            const float* __restrict__ bout,
                            float* __restrict__ out) {                // [B*L][VOCAB]
    const int lane = threadIdx.x & 31;
    const int wave = threadIdx.x >> 5;
    const long flat = (long)blockIdx.x * (PROJ_THREADS >> 5) + wave;
    if (flat >= (long)MTILES * NGROUPS) return;
    const int mtile = (int)(flat / NGROUPS);
    const int ngrp  = (int)(flat % NGROUPS);
    const int m0 = mtile * 16;
    const int n0 = ngrp * (16 * NT_PER_WAVE);

    const int arow = lane & 15;
    const int koff = (lane >> 4) * 8;

    FragC acc[NT_PER_WAVE];
    #pragma unroll
    for (int j = 0; j < NT_PER_WAVE; ++j)
        #pragma unroll
        for (int i = 0; i < 8; ++i) acc[j].f[i] = 0.0f;

    const unsigned short* Arow = ys + (size_t)(m0 + arow) * D_;
    const unsigned short* Brow[NT_PER_WAVE];
    #pragma unroll
    for (int j = 0; j < NT_PER_WAVE; ++j)
        Brow[j] = Woutb + (size_t)(n0 + j * 16 + (lane & 15)) * D_;

    // 1-deep software pipeline over the K loop.
    FragAB fa, fb[NT_PER_WAVE];
    fa.q[0] = *(const v4u*)(Arow + koff);
    fa.q[1] = *(const v4u*)(Arow + 16 + koff);
    #pragma unroll
    for (int j = 0; j < NT_PER_WAVE; ++j) {
        fb[j].q[0] = *(const v4u*)(Brow[j] + koff);
        fb[j].q[1] = *(const v4u*)(Brow[j] + 16 + koff);
    }

    #pragma unroll
    for (int k0 = 0; k0 < D_; k0 += 32) {
        const int kn = k0 + 32;
        FragAB nfa, nfb[NT_PER_WAVE];
        if (kn < D_) {
            nfa.q[0] = *(const v4u*)(Arow + kn + koff);
            nfa.q[1] = *(const v4u*)(Arow + kn + 16 + koff);
            #pragma unroll
            for (int j = 0; j < NT_PER_WAVE; ++j) {
                nfb[j].q[0] = *(const v4u*)(Brow[j] + kn + koff);
                nfb[j].q[1] = *(const v4u*)(Brow[j] + kn + 16 + koff);
            }
        }
        #pragma unroll
        for (int j = 0; j < NT_PER_WAVE; ++j) {
            acc[j].v = __builtin_amdgcn_wmma_f32_16x16x32_bf16(
                false, fa.v, false, fb[j].v, (short)0, acc[j].v, false, false);
        }
        if (kn < D_) {
            fa = nfa;
            #pragma unroll
            for (int j = 0; j < NT_PER_WAVE; ++j) fb[j] = nfb[j];
        }
    }

    const int rbase = m0 + ((lane >> 4) * 8);   // C rows: VGPR idx (+8 for hi lanes)
    #pragma unroll
    for (int j = 0; j < NT_PER_WAVE; ++j) {
        const int n = n0 + j * 16 + (lane & 15);
        const float bn = bout[n];
        #pragma unroll
        for (int r = 0; r < 8; ++r) {
            out[(size_t)(rbase + r) * VOCAB + n] = acc[j].f[r] + bn;
        }
    }
}

extern "C" void kernel_launch(void* const* d_in, const int* in_sizes, int n_in,
                              void* d_out, int out_size, void* d_ws, size_t ws_size,
                              hipStream_t stream) {
    const int*   ids     = (const int*)  d_in[0];
    const float* embed   = (const float*)d_in[1];
    const float* starter = (const float*)d_in[2];
    const float* W       = (const float*)d_in[3];
    const float* bias    = (const float*)d_in[4];
    const float* Wout    = (const float*)d_in[5];
    const float* bout    = (const float*)d_in[6];
    float* out = (float*)d_out;

    char* ws = (char*)d_ws;
    size_t off = 0;
    auto alloc = [&](size_t bytes) -> void* {
        void* p = ws + off;
        off += (bytes + 255) & ~(size_t)255;
        return p;
    };
    unsigned*       counter = (unsigned*)alloc(256);
    unsigned short* Wb      = (unsigned short*)alloc((size_t)NLAYERS * TWOD * TWOD * 2);
    unsigned short* Woutb   = (unsigned short*)alloc((size_t)VOCAB * D_ * 2);
    unsigned short* ys      = (unsigned short*)alloc((size_t)B_ * L_ * D_ * 2);
    float*          Xbuf    = (float*)alloc((size_t)2 * B_ * D_ * 4);
    float*          Sbuf    = (float*)alloc((size_t)2 * NLAYERS * B_ * D_ * 4);
    (void)ws_size; (void)in_sizes; (void)n_in; (void)out_size;

    hipMemsetAsync(counter, 0, 256, stream);

    const int nW = NLAYERS * TWOD * TWOD;
    cvt_f32_to_bf16<<<(nW + 255) / 256, 256, 0, stream>>>(W, Wb, nW);
    const int nWo = VOCAB * D_;
    cvt_f32_to_bf16<<<(nWo + 255) / 256, 256, 0, stream>>>(Wout, Woutb, nWo);
    const int nS = NLAYERS * B_ * D_;
    init_state_kernel<<<(nS + 255) / 256, 256, 0, stream>>>(starter, Sbuf);

    rnn_kernel<<<RNN_NWG, RNN_THREADS, 0, stream>>>(
        ids, embed, bias, Wb, Xbuf, Sbuf, ys, counter);

    const long total_waves = (long)MTILES * NGROUPS;          // 128000
    const int  nblocks = (int)((total_waves + (PROJ_THREADS / 32) - 1) / (PROJ_THREADS / 32));
    proj_kernel<<<nblocks, PROJ_THREADS, 0, stream>>>(ys, Woutb, bout, out);
}
